// PolyHashV11_87016037416975
// MI455X (gfx1250) — matrix operand order repaired and checked by reading, not compiled, not run on hardware
//
#include <hip/hip_runtime.h>
#include <hip/hip_bf16.h>
#include <math.h>

typedef __attribute__((ext_vector_type(16))) _Float16 v16h;
typedef __attribute__((ext_vector_type(8)))  _Float16 v8h;
typedef __attribute__((ext_vector_type(8)))  float    v8f;

#define BUCKETS 2000000ULL

__constant__ unsigned long long PRIMES_C[8] = {
    2654435761ULL, 2246822519ULL, 3266489917ULL, 2028178513ULL,
    1220703125ULL, 1610612741ULL,  805306457ULL,  402653189ULL};

__device__ __forceinline__ float silu_f(float x) {
    return x / (1.0f + __expf(-x));
}

// Force GLOBAL (addrspace 1) vector loads so the backend emits
// global_load_b128 (LOADcnt only) instead of flat_load_b128 (LOADcnt+DScnt,
// ties up the LDS datapath per CDNA5 ISA 10.1.1).
__device__ __forceinline__ v8h load_global_v8h(const _Float16* p) {
    return *(const __attribute__((address_space(1))) v8h*)(unsigned long long)p;
}
__device__ __forceinline__ v16h load_global_v16h(const _Float16* p) {
    return *(const __attribute__((address_space(1))) v16h*)(unsigned long long)p;
}

// ---------------------------------------------------------------------------
// Weight f32[N,K] -> f16[N,Kp] (zero padded in K)
// ---------------------------------------------------------------------------
__global__ void cvt_weight_kernel(const float* __restrict__ src, _Float16* __restrict__ dst,
                                  int N, int K, int Kp) {
    int idx = blockIdx.x * blockDim.x + threadIdx.x;
    if (idx >= N * Kp) return;
    int n = idx / Kp, k = idx - n * Kp;
    dst[idx] = (k < K) ? (_Float16)src[n * K + k] : (_Float16)0.0f;
}

// ---------------------------------------------------------------------------
// Hash features: rolling polynomial hashes, table gathers, conditioned keys.
// Writes hfeat_raw [B,T,8] f32.
// ---------------------------------------------------------------------------
__global__ void features_kernel(const int* __restrict__ tok,
                                const float* __restrict__ t0, const float* __restrict__ t1,
                                const float* __restrict__ t2, const float* __restrict__ t3,
                                const float* __restrict__ cond_w,
                                float* __restrict__ hfeat, int B, int T) {
    int idx = blockIdx.x * blockDim.x + threadIdx.x;
    if (idx >= B * T) return;
    int b = idx / T, t = idx - b * T;
    const int* row = tok + (size_t)b * T;

    unsigned long long term[8];
#pragma unroll
    for (int i = 0; i < 8; i++) {
        int ti = t - 1 - i;
        unsigned long long tv = (ti >= 0) ? (unsigned long long)(unsigned)row[ti] : 0ULL;
        term[i] = tv * PRIMES_C[i];
    }
    unsigned long long h = term[0];
    unsigned long long k0 = h % BUCKETS;
    h ^= term[1];
    unsigned long long k1 = h % BUCKETS;
    h ^= term[2]; h ^= term[3];
    unsigned long long k2 = h % BUCKETS;
    h ^= term[4]; h ^= term[5]; h ^= term[6]; h ^= term[7];
    unsigned long long k3 = h % BUCKETS;

    float s00 = t0[k0 * 2], s01 = t0[k0 * 2 + 1];
    float s10 = t1[k1 * 2], s11 = t1[k1 * 2 + 1];

    unsigned long long ck = 0ULL;
#pragma unroll
    for (int j = 0; j < 8; j++) {
        float lg = cond_w[j * 4 + 0] * s00 + cond_w[j * 4 + 1] * s01 +
                   cond_w[j * 4 + 2] * s10 + cond_w[j * 4 + 3] * s11;
        if (lg > 0.0f) ck ^= PRIMES_C[j];
    }
    unsigned long long k2c = (k2 ^ ck) % BUCKETS;
    unsigned long long k3c = (k3 ^ ck) % BUCKETS;

    float* o = hfeat + (size_t)idx * 8;
    o[0] = s00; o[1] = s01; o[2] = s10; o[3] = s11;
    o[4] = t2[k2c * 2]; o[5] = t2[k2c * 2 + 1];
    o[6] = t3[k3c * 2]; o[7] = t3[k3c * 2 + 1];
}

// ---------------------------------------------------------------------------
// Assemble h = [byte_emb(64) | silu(causal depthwise conv of hfeat)(8) | mf(8)]
// Writes f32 [B,T,80] and f16 [B,T,96] (zero padded).
// ---------------------------------------------------------------------------
__global__ void assemble_kernel(const int* __restrict__ tok, const float* __restrict__ hfeat,
                                const float* __restrict__ emb,
                                const float* __restrict__ cw, const float* __restrict__ cb,
                                float* __restrict__ h80, _Float16* __restrict__ h16,
                                int B, int T) {
    int idx = blockIdx.x * blockDim.x + threadIdx.x;
    if (idx >= B * T) return;
    int b = idx / T, t = idx - b * T;
    const int* row = tok + (size_t)b * T;
    int tv = row[t];

    float*     o32 = h80 + (size_t)idx * 80;
    _Float16*  o16 = h16 + (size_t)idx * 96;
    const float* e = emb + (size_t)tv * 64;
#pragma unroll 8
    for (int i = 0; i < 64; i++) {
        float v = e[i];
        o32[i] = v; o16[i] = (_Float16)v;
    }
#pragma unroll
    for (int c = 0; c < 8; c++) {
        float acc = cb[c];
#pragma unroll
        for (int j = 0; j < 8; j++) {
            int ts = t - 7 + j;
            if (ts >= 0) acc += cw[c * 8 + j] * hfeat[((size_t)b * T + ts) * 8 + c];
        }
        float v = silu_f(acc);
        o32[64 + c] = v; o16[64 + c] = (_Float16)v;
    }
#pragma unroll
    for (int k = 1; k <= 8; k++) {
        float m = (t >= k && row[t - k] == tv) ? 1.0f : 0.0f;
        o32[71 + k] = m; o16[71 + k] = (_Float16)m;
    }
#pragma unroll
    for (int i = 80; i < 96; i++) o16[i] = (_Float16)0.0f;
}

// ---------------------------------------------------------------------------
// Gated linear recurrence: h_t = g*h_{t-1} + (1-g)*u, g = sigmoid(zg).
// One thread per (b,s). Writes states f16 [B,T,S].
// ---------------------------------------------------------------------------
__global__ void scan_kernel(const float* __restrict__ zg, const float* __restrict__ zi,
                            _Float16* __restrict__ st16, int B, int T, int S) {
    int idx = blockIdx.x * blockDim.x + threadIdx.x;
    if (idx >= B * S) return;
    int b = idx / S, s = idx - b * S;
    float h = 0.0f;
    size_t base = (size_t)b * T * S + s;
    for (int t = 0; t < T; t++) {
        size_t o = base + (size_t)t * S;
        float g = 1.0f / (1.0f + __expf(-zg[o]));
        h = g * h + (1.0f - g) * zi[o];
        st16[o] = (_Float16)h;
    }
}

// ---------------------------------------------------------------------------
// LayerNorm over 80 of (x + y), writes f16 [rows,96] zero padded. 1 wave/row.
// ---------------------------------------------------------------------------
__global__ __launch_bounds__(256) void ln80_kernel(const float* __restrict__ x,
                                                   const float* __restrict__ y,
                                                   const float* __restrict__ w,
                                                   const float* __restrict__ bi,
                                                   _Float16* __restrict__ out, int rows) {
    int row = blockIdx.x * 8 + (threadIdx.x >> 5);
    if (row >= rows) return;
    int lane = threadIdx.x & 31;
    const float* xr = x + (size_t)row * 80;
    const float* yr = y + (size_t)row * 80;
    float v0 = xr[lane] + yr[lane];
    float v1 = xr[lane + 32] + yr[lane + 32];
    float v2 = (lane < 16) ? (xr[lane + 64] + yr[lane + 64]) : 0.0f;
    float sum = v0 + v1 + v2;
    float sq  = v0 * v0 + v1 * v1 + v2 * v2;
#pragma unroll
    for (int off = 16; off; off >>= 1) {
        sum += __shfl_xor(sum, off, 32);
        sq  += __shfl_xor(sq,  off, 32);
    }
    float mean = sum * (1.0f / 80.0f);
    float var  = sq * (1.0f / 80.0f) - mean * mean;
    float rstd = rsqrtf(var + 1e-5f);
    _Float16* o = out + (size_t)row * 96;
    o[lane]      = (_Float16)((v0 - mean) * rstd * w[lane] + bi[lane]);
    o[lane + 32] = (_Float16)((v1 - mean) * rstd * w[lane + 32] + bi[lane + 32]);
    if (lane < 16) o[lane + 64] = (_Float16)((v2 - mean) * rstd * w[lane + 64] + bi[lane + 64]);
    else           o[lane + 64] = (_Float16)0.0f;   // pad 80..95
}

// ---------------------------------------------------------------------------
// LayerNorm over 256 of (a + r), writes f16 [rows,256]. 1 wave/row.
// ---------------------------------------------------------------------------
__global__ __launch_bounds__(256) void ln256_kernel(const float* __restrict__ a,
                                                    const float* __restrict__ r,
                                                    const float* __restrict__ w,
                                                    const float* __restrict__ bi,
                                                    _Float16* __restrict__ out, int rows) {
    int row = blockIdx.x * 8 + (threadIdx.x >> 5);
    if (row >= rows) return;
    int lane = threadIdx.x & 31;
    const float* ar = a + (size_t)row * 256;
    const float* rr = r + (size_t)row * 256;
    float v[8], sum = 0.0f, sq = 0.0f;
#pragma unroll
    for (int j = 0; j < 8; j++) {
        float x = ar[lane + 32 * j] + rr[lane + 32 * j];
        v[j] = x; sum += x; sq += x * x;
    }
#pragma unroll
    for (int off = 16; off; off >>= 1) {
        sum += __shfl_xor(sum, off, 32);
        sq  += __shfl_xor(sq,  off, 32);
    }
    float mean = sum * (1.0f / 256.0f);
    float var  = sq * (1.0f / 256.0f) - mean * mean;
    float rstd = rsqrtf(var + 1e-5f);
    _Float16* o = out + (size_t)row * 256;
#pragma unroll
    for (int j = 0; j < 8; j++) {
        int i = lane + 32 * j;
        o[i] = (_Float16)((v[j] - mean) * rstd * w[i] + bi[i]);
    }
}

// ---------------------------------------------------------------------------
// Elementwise product p = a1 * a2 (f16), in place into a1.
// ---------------------------------------------------------------------------
__global__ void prod_kernel(_Float16* __restrict__ a1, const _Float16* __restrict__ a2, int n) {
    int idx = blockIdx.x * blockDim.x + threadIdx.x;
    if (idx >= n) return;
    a1[idx] = (_Float16)((float)a1[idx] * (float)a2[idx]);
}

// ---------------------------------------------------------------------------
// WMMA GEMM, ping-pong software pipeline (no register rotation copies).
// out[m,n] = act(sum_k A[m,k]*W[n,k] + bias[n])
// A f16 [M,lda], W f16 [N,ldw] row-major (K contiguous -> B fragment in contiguous
// k order k = 16*(lane/16) + e, as A: lane = N column, K striped across v16h elements).
// Template NT = number of 16-wide n-subtiles per wave; N must be a multiple
// of 16*NT per launch. Wave tile = 16 x (16*NT); block = 8 waves = 128 rows.
// K multiple of 32; M multiple of 128 (always 65536 here).
// ---------------------------------------------------------------------------
template <int NT>
__device__ __forceinline__ void gemm_load_frags(const _Float16* aptr,
                                                const _Float16* const* wptr,
                                                v8h& lo, v8h& hi, v16h* bf) {
    // A fragment, 16-bit A 16x32 layout (contiguous, matching B):
    //  lanes 0-15 : e0-15 -> K +0..15
    //  lanes 16-31: e0-15 -> K +16..31
    lo = load_global_v8h(aptr);
    hi = load_global_v8h(aptr + 8);
#pragma unroll
    for (int i = 0; i < NT; i++)
        bf[i] = load_global_v16h(wptr[i]);   // B 32x16: contiguous 32B per lane
}

template <int NT>
__device__ __forceinline__ void gemm_wmma_step(const v8h& lo, const v8h& hi,
                                               const v16h* bf, v8f* acc) {
    v16h afrag = __builtin_shufflevector(lo, hi, 0, 1, 2, 3, 4, 5, 6, 7,
                                         8, 9, 10, 11, 12, 13, 14, 15);
#pragma unroll
    for (int i = 0; i < NT; i++)
        acc[i] = __builtin_amdgcn_wmma_f32_16x16x32_f16(
            false, afrag, false, bf[i], (short)0, acc[i], false, false);
}

template <int NT>
__global__ __launch_bounds__(256) void wmma_gemm_kernel(
        const _Float16* __restrict__ A, int lda,
        const _Float16* __restrict__ W, int ldw,
        const float* __restrict__ bias,
        float* __restrict__ out32, int ldo32,
        _Float16* __restrict__ out16, int ldo16,
        int K, int act) {
    const int wave  = threadIdx.x >> 5;
    const int lane  = threadIdx.x & 31;
    const int grp   = lane >> 4;           // 0/1: K-half selector
    const int l15   = lane & 15;
    const int mTile = (blockIdx.y * 8 + wave) * 16;
    const int nBase = blockIdx.x * (16 * NT);

    v8f acc[NT];
#pragma unroll
    for (int i = 0; i < NT; i++)
#pragma unroll
        for (int j = 0; j < 8; j++) acc[i][j] = 0.0f;

    // Hoisted per-lane pointers, bumped by 32 halves per K-step.
    const _Float16* aptr = A + (size_t)(mTile + l15) * lda + grp * 16;
    const _Float16* wptr[NT];
#pragma unroll
    for (int i = 0; i < NT; i++)
        wptr[i] = W + (size_t)(nBase + i * 16 + l15) * ldw + grp * 16;

    const int steps = K >> 5;   // >= 1

    // Ping-pong double buffering: buf0/buf1 alternate, no rotation copies.
    v8h lo0, hi0; v16h b0[NT];
    gemm_load_frags<NT>(aptr, wptr, lo0, hi0, b0);

    int s = 0;
    for (; s + 2 <= steps; s += 2) {
        // Loads for step s+1 into buf1.
        aptr += 32;
#pragma unroll
        for (int i = 0; i < NT; i++) wptr[i] += 32;
        v8h lo1, hi1; v16h b1[NT];
        gemm_load_frags<NT>(aptr, wptr, lo1, hi1, b1);

        // Compute step s from buf0 (buf1 loads stay outstanding).
        gemm_wmma_step<NT>(lo0, hi0, b0, acc);

        // Refill buf0 with step s+2 (uniform scalar guard).
        if (s + 2 < steps) {
            aptr += 32;
#pragma unroll
            for (int i = 0; i < NT; i++) wptr[i] += 32;
            gemm_load_frags<NT>(aptr, wptr, lo0, hi0, b0);
        }

        // Compute step s+1 from buf1.
        gemm_wmma_step<NT>(lo1, hi1, b1, acc);
    }
    if (s < steps) {
        // Odd leftover step lives in buf0.
        gemm_wmma_step<NT>(lo0, hi0, b0, acc);
    }

    // C/D f32 16x16 layout: lane n = l15; VGPR v -> m = v + grp*8
#pragma unroll
    for (int i = 0; i < NT; i++) {
        int n = nBase + i * 16 + l15;
        float bv = bias ? bias[n] : 0.0f;
#pragma unroll
        for (int v = 0; v < 8; v++) {
            int m = mTile + v + grp * 8;
            float val = acc[i][v] + bv;
            if (act == 1) val = silu_f(val);
            if (out32) out32[(size_t)m * ldo32 + n] = val;
            if (out16) out16[(size_t)m * ldo16 + n] = (_Float16)val;
        }
    }
}

// ---------------------------------------------------------------------------
// Host side
// ---------------------------------------------------------------------------
static inline void launch_gemm(hipStream_t s, const _Float16* A, int lda,
                               const _Float16* W, int ldw, const float* bias,
                               float* o32, int l32, _Float16* o16, int l16,
                               int M, int N, int K, int act) {
    if (N % 64 == 0) {
        dim3 grid(N / 64, M / 128);
        wmma_gemm_kernel<4><<<grid, 256, 0, s>>>(A, lda, W, ldw, bias, o32, l32, o16, l16, K, act);
    } else if (N % 80 == 0) {
        dim3 grid(N / 80, M / 128);
        wmma_gemm_kernel<5><<<grid, 256, 0, s>>>(A, lda, W, ldw, bias, o32, l32, o16, l16, K, act);
    } else { // N % 32 == 0
        dim3 grid(N / 32, M / 128);
        wmma_gemm_kernel<2><<<grid, 256, 0, s>>>(A, lda, W, ldw, bias, o32, l32, o16, l16, K, act);
    }
}

static inline void launch_cvt(hipStream_t s, const float* src, _Float16* dst, int N, int K, int Kp) {
    int tot = N * Kp;
    cvt_weight_kernel<<<(tot + 255) / 256, 256, 0, s>>>(src, dst, N, K, Kp);
}

extern "C" void kernel_launch(void* const* d_in, const int* in_sizes, int n_in,
                              void* d_out, int out_size, void* d_ws, size_t ws_size,
                              hipStream_t stream) {
    const int B = 32, T = 2048, S = 32;
    const int R = B * T;   // 65536 rows

    const int*   chars     = (const int*)  d_in[0];
    const float* byte_emb  = (const float*)d_in[1];
    const float* table0    = (const float*)d_in[2];
    const float* table1    = (const float*)d_in[3];
    const float* table2    = (const float*)d_in[4];
    const float* table3    = (const float*)d_in[5];
    const float* cond_w    = (const float*)d_in[6];
    const float* conv_w    = (const float*)d_in[7];
    const float* conv_b    = (const float*)d_in[8];
    const float* gate_w    = (const float*)d_in[9];
    const float* gate_b    = (const float*)d_in[10];
    const float* inpr_w    = (const float*)d_in[11];
    const float* inpr_b    = (const float*)d_in[12];
    const float* outpr_w   = (const float*)d_in[13];
    const float* outpr_b   = (const float*)d_in[14];
    const float* scan_ln_w = (const float*)d_in[15];
    const float* scan_ln_b = (const float*)d_in[16];
    const float* inp_w     = (const float*)d_in[17];
    const float* inp_b     = (const float*)d_in[18];
    const float* r_w1      = (const float*)d_in[19];
    const float* r_w2      = (const float*)d_in[20];
    const float* r_proj    = (const float*)d_in[21];
    const float* r_ln_w    = (const float*)d_in[22];
    const float* r_ln_b    = (const float*)d_in[23];
    const float* out_w     = (const float*)d_in[24];
    const float* out_b     = (const float*)d_in[25];
    float* outp = (float*)d_out;

    // ---- workspace carve (256B aligned) ----
    char* ws = (char*)d_ws;
    size_t off = 0;
    auto carve = [&](size_t bytes) -> char* {
        char* p = ws + off;
        off = (off + bytes + 255) & ~(size_t)255;
        return p;
    };
    _Float16* w16_gate  = (_Float16*)carve(32   * 96  * 2);
    _Float16* w16_inpr  = (_Float16*)carve(32   * 96  * 2);
    _Float16* w16_outpr = (_Float16*)carve(80   * 32  * 2);
    _Float16* w16_inp   = (_Float16*)carve(256  * 96  * 2);
    _Float16* w16_r1    = (_Float16*)carve(256  * 256 * 2);
    _Float16* w16_r2    = (_Float16*)carve(256  * 256 * 2);
    _Float16* w16_rproj = (_Float16*)carve(256  * 256 * 2);
    _Float16* w16_out   = (_Float16*)carve(1024 * 256 * 2);
    float*    hfeat     = (float*)   carve((size_t)R * 8   * 4);
    float*    h80f32    = (float*)   carve((size_t)R * 80  * 4);
    _Float16* h80f16    = (_Float16*)carve((size_t)R * 96  * 2);
    float*    zg        = (float*)   carve((size_t)R * 32  * 4);
    float*    zi        = (float*)   carve((size_t)R * 32  * 4);
    _Float16* st16      = (_Float16*)carve((size_t)R * 32  * 2);
    float*    y80       = (float*)   carve((size_t)R * 80  * 4);
    _Float16* hpost16   = (_Float16*)carve((size_t)R * 96  * 2);
    float*    h256f32   = (float*)   carve((size_t)R * 256 * 4);
    _Float16* h256f16   = (_Float16*)carve((size_t)R * 256 * 2);
    _Float16* a1        = (_Float16*)carve((size_t)R * 256 * 2);
    _Float16* a2        = (_Float16*)carve((size_t)R * 256 * 2);
    float*    hh        = (float*)   carve((size_t)R * 256 * 4);
    _Float16* h2f16     = (_Float16*)carve((size_t)R * 256 * 2);
    (void)ws_size; (void)in_sizes; (void)n_in; (void)out_size;

    // ---- weight conversion (f32 [N,K] -> f16 [N,Kp]) ----
    launch_cvt(stream, gate_w,  w16_gate,  32,   80,  96);
    launch_cvt(stream, inpr_w,  w16_inpr,  32,   80,  96);
    launch_cvt(stream, outpr_w, w16_outpr, 80,   32,  32);
    launch_cvt(stream, inp_w,   w16_inp,   256,  80,  96);
    launch_cvt(stream, r_w1,    w16_r1,    256,  256, 256);
    launch_cvt(stream, r_w2,    w16_r2,    256,  256, 256);
    launch_cvt(stream, r_proj,  w16_rproj, 256,  256, 256);
    launch_cvt(stream, out_w,   w16_out,   1024, 256, 256);

    // ---- features + assemble ----
    features_kernel<<<(R + 255) / 256, 256, 0, stream>>>(
        chars, table0, table1, table2, table3, cond_w, hfeat, B, T);
    assemble_kernel<<<(R + 255) / 256, 256, 0, stream>>>(
        chars, hfeat, byte_emb, conv_w, conv_b, h80f32, h80f16, B, T);

    // ---- scan projections (WMMA) + recurrence ----
    launch_gemm(stream, h80f16, 96, w16_gate, 96, gate_b, zg, 32, nullptr, 0, R, 32, 96, 0);
    launch_gemm(stream, h80f16, 96, w16_inpr, 96, inpr_b, zi, 32, nullptr, 0, R, 32, 96, 0);
    scan_kernel<<<(B * S + 255) / 256, 256, 0, stream>>>(zg, zi, st16, B, T, S);
    launch_gemm(stream, st16, 32, w16_outpr, 32, outpr_b, y80, 80, nullptr, 0, R, 80, 32, 0);
    ln80_kernel<<<R / 8, 256, 0, stream>>>(h80f32, y80, scan_ln_w, scan_ln_b, hpost16, R);

    // ---- MLP in: silu(h @ inp_w^T + b), f32 + f16 outputs ----
    launch_gemm(stream, hpost16, 96, w16_inp, 96, inp_b, h256f32, 256, h256f16, 256, R, 256, 96, 1);

    // ---- gated block: silu(h@r_w1^T) * (h@r_w2^T) @ r_proj^T ----
    launch_gemm(stream, h256f16, 256, w16_r1, 256, nullptr, nullptr, 0, a1, 256, R, 256, 256, 1);
    launch_gemm(stream, h256f16, 256, w16_r2, 256, nullptr, nullptr, 0, a2, 256, R, 256, 256, 0);
    prod_kernel<<<((R * 256) + 255) / 256, 256, 0, stream>>>(a1, a2, R * 256);
    launch_gemm(stream, a1, 256, w16_rproj, 256, nullptr, hh, 256, nullptr, 0, R, 256, 256, 0);
    ln256_kernel<<<R / 8, 256, 0, stream>>>(hh, h256f32, r_ln_w, r_ln_b, h2f16, R);

    // ---- LM head -> d_out [R,1024] f32 ----
    launch_gemm(stream, h2f16, 256, w16_out, 256, out_b, outp, 1024, nullptr, 0, R, 1024, 256, 0);
}
